// ConsensusAttnResUNetStudent_11828339933526
// MI455X (gfx1250) — matrix operand order repaired and checked by reading, not compiled, not run on hardware
//
#include <hip/hip_runtime.h>
#include <hip/hip_bf16.h>
#include <cstdint>

// ---------------------------------------------------------------------------
// CDNA5 / gfx1250 WMMA helpers (wave32).
// ---------------------------------------------------------------------------
typedef __attribute__((ext_vector_type(16))) _Float16 v16h;
typedef __attribute__((ext_vector_type(8)))  _Float16 h8;
typedef __attribute__((ext_vector_type(8)))  float    v8f;
typedef __attribute__((ext_vector_type(4)))  float    f4;

#define DEV __device__ __forceinline__

DEV v8f wmma16x16x32(v16h a, v16h b, v8f c) {
  // D(f32 16x16) = A(f16 16x32) * B(f16 32x16) + C
  return __builtin_amdgcn_wmma_f32_16x16x32_f16(
      /*neg_a=*/false, a, /*neg_b=*/false, b,
      /*c_mod=*/(short)0, c, /*reuse_a=*/false, /*reuse_b=*/false);
}

DEV h8 ld8(const _Float16* p) { return *(const h8*)p; }
DEV void st8(_Float16* p, h8 v) { *(h8*)p = v; }
DEV v16h cat16(h8 lo, h8 hi) {
  v16h r;
#pragma unroll
  for (int e = 0; e < 8; ++e) { r[e] = lo[e]; r[e + 8] = hi[e]; }
  return r;
}
// A operand (16x32 f16): lane row = lane&15; elems 0-7 = K kk+8*half..+7,
// elems 8-15 = K kk+16+8*half..+7  -> two contiguous b128 loads.
DEV v16h loadA(const _Float16* row, int kk, int half) {
  return cat16(ld8(row + kk + 8 * half), ld8(row + kk + 16 + 8 * half));
}
// B operand (32x16 f16): lane col = lane&15; elems 0-7 = K kk+8*half..+7,
// elems 8-15 = K kk+16+8*half..+7 -> two contiguous b128 loads.
DEV v16h loadB(const _Float16* col, int kk, int half) {
  return cat16(ld8(col + kk + 8 * half), ld8(col + kk + 16 + 8 * half));
}
DEV v8f vzero() { v8f z; for (int i = 0; i < 8; ++i) z[i] = 0.f; return z; }

// Problem constants
constexpr int Bn    = 2;
constexpr int Cc    = 512;
constexpr int HEADS = 2;
constexpr int Dd    = 256;        // head dim
constexpr int NPIX  = 64;         // 8x8 image
constexpr int NWIN  = 64;         // 8x8 window grid (stride 1, reflect-pad 15x15)
constexpr int NTOK  = 64;         // 8x8 window tokens
constexpr int Lg    = NWIN * NTOK;// 4096 global tokens

// reflect-pad (3 top/left, 4 bottom/right): window token -> source pixel
DEV int tok_src(int win, int t) {
  int r = (win >> 3) + (t >> 3) - 3;
  int c = (win & 7)  + (t & 7)  - 3;
  r = r < 0 ? -r : (r > 7 ? 14 - r : r);
  c = c < 0 ? -c : (c > 7 ? 14 - c : c);
  return r * 8 + c;
}

// ---------------------------------------------------------------------------
// K0: f32 [b][c][p] -> f16 pixel-major [b][p][c] (K-contiguous for WMMA B).
// ---------------------------------------------------------------------------
__global__ void to_f16_tr_kernel(const float* __restrict__ x, _Float16* __restrict__ xt) {
  int idx = blockIdx.x * blockDim.x + threadIdx.x;     // b*C*64 total
  int b = idx / (Cc * NPIX);
  int rem = idx % (Cc * NPIX);
  int c = rem / NPIX, p = rem % NPIX;
  xt[((size_t)b * NPIX + p) * Cc + c] = (_Float16)x[idx];
}

// ---------------------------------------------------------------------------
// K1: projection GEMM: out[b][p][o] (f16) = sum_c W[o,c]*x[b,c,p] + bias[o]
// ---------------------------------------------------------------------------
__global__ void proj_gemm_kernel(const float* __restrict__ W,
                                 const float* __restrict__ bias,
                                 const _Float16* __restrict__ xt,
                                 _Float16* __restrict__ out, int O) {
  const int tid = threadIdx.x;
  const int wv = tid >> 5, lane = tid & 31, half = lane >> 4, ln = lane & 15;
  const int b = blockIdx.y, mt = blockIdx.x;
  const int o = mt * 16 + ln;                 // A row
  const int p = wv * 16 + ln;                 // B / D col
  const float*    wr = W  + (size_t)o * Cc;
  const _Float16* xr = xt + ((size_t)b * NPIX + p) * Cc;

  v8f acc = vzero();
  for (int kk = 0; kk < Cc; kk += 32) {
    f4 w0 = *(const f4*)(wr + kk + 8 * half);
    f4 w1 = *(const f4*)(wr + kk + 8 * half + 4);
    f4 w2 = *(const f4*)(wr + kk + 16 + 8 * half);
    f4 w3 = *(const f4*)(wr + kk + 20 + 8 * half);
    v16h a;
#pragma unroll
    for (int e = 0; e < 4; ++e) {
      a[e]      = (_Float16)w0[e];
      a[e + 4]  = (_Float16)w1[e];
      a[e + 8]  = (_Float16)w2[e];
      a[e + 12] = (_Float16)w3[e];
    }
    acc = wmma16x16x32(a, loadB(xr, kk, half), acc);
  }
  h8 o8;
#pragma unroll
  for (int r = 0; r < 8; ++r) o8[r] = (_Float16)(acc[r] + bias[mt * 16 + 8 * half + r]);
  st8(out + ((size_t)b * NPIX + p) * O + mt * 16 + 8 * half, o8);  // 8 D-rows contiguous
}

// ---------------------------------------------------------------------------
// K2: fused local window attention; one WG (4 waves) per (b, win, head).
// ---------------------------------------------------------------------------
__global__ void local_attn_kernel(const _Float16* __restrict__ qkvx,  // [b][pix][3C]
                                  const _Float16* __restrict__ qkz,   // [b][pix][2C]
                                  _Float16* __restrict__ qg,          // [bh][l][d]
                                  _Float16* __restrict__ qgT,         // [bh][d][l]
                                  const float* __restrict__ pbeta) {
  __shared__ _Float16 sVt[Dd * NTOK];   // 32 KB: V tile transposed [dim][tok]
  __shared__ _Float16 sP[NTOK * NTOK];  //  8 KB: probs
  const int tid = threadIdx.x;
  const int wv = tid >> 5, lane = tid & 31, half = lane >> 4, ln = lane & 15;
  const int win = blockIdx.x, head = blockIdx.y, b = blockIdx.z;
  const float beta  = log1pf(__expf(pbeta[0])) + 1e-6f;   // softplus + eps
  const float scale = 0.0625f;                            // 1/sqrt(256)
  const int co = head * Dd;
  const _Float16* xb = qkvx + (size_t)b * NPIX * (3 * Cc);
  const _Float16* zb = qkz  + (size_t)b * NPIX * (2 * Cc);

  // stage V transposed into LDS (vector global reads, scalar LDS scatter)
  for (int i = tid; i < NTOK * (Dd / 8); i += 128) {
    int tok = i >> 5, dc = i & 31;
    h8 v = ld8(xb + (size_t)tok_src(win, tok) * (3 * Cc) + 2 * Cc + co + dc * 8);
#pragma unroll
    for (int j = 0; j < 8; ++j) sVt[(dc * 8 + j) * NTOK + tok] = v[j];
  }
  __syncthreads();

  const _Float16* qxr = xb + (size_t)tok_src(win, wv * 16 + ln) * (3 * Cc) + co;
  const _Float16* qzr = zb + (size_t)tok_src(win, wv * 16 + ln) * (2 * Cc) + co;
  int pixB[4];
#pragma unroll
  for (int nt = 0; nt < 4; ++nt) pixB[nt] = tok_src(win, nt * 16 + ln);

  v8f cx[4], cz[4];
  for (int nt = 0; nt < 4; ++nt) { cx[nt] = vzero(); cz[nt] = vzero(); }

  for (int kk = 0; kk < Dd; kk += 32) {
    v16h ax = loadA(qxr, kk, half);
    v16h az = loadA(qzr, kk, half);
#pragma unroll
    for (int nt = 0; nt < 4; ++nt) {
      const _Float16* kxr = xb + (size_t)pixB[nt] * (3 * Cc) + Cc + co;
      const _Float16* kzr = zb + (size_t)pixB[nt] * (2 * Cc) + Cc + co;
      cx[nt] = wmma16x16x32(ax, loadB(kxr, kk, half), cx[nt]);
      cz[nt] = wmma16x16x32(az, loadB(kzr, kk, half), cz[nt]);
    }
  }

  // softmax over 64 cols; lane owns rows m=r+8*half at col nt*16+ln
  float lt[4][8];
#pragma unroll
  for (int nt = 0; nt < 4; ++nt)
#pragma unroll
    for (int r = 0; r < 8; ++r) lt[nt][r] = scale * (cx[nt][r] + beta * cz[nt][r]);

#pragma unroll
  for (int r = 0; r < 8; ++r) {
    float mx = lt[0][r];
    for (int nt = 1; nt < 4; ++nt) mx = fmaxf(mx, lt[nt][r]);
    for (int off = 1; off < 16; off <<= 1) mx = fmaxf(mx, __shfl_xor(mx, off, 32));
    float s = 0.f;
    for (int nt = 0; nt < 4; ++nt) { lt[nt][r] = __expf(lt[nt][r] - mx); s += lt[nt][r]; }
    for (int off = 1; off < 16; off <<= 1) s += __shfl_xor(s, off, 32);
    float inv = 1.0f / s;
    for (int nt = 0; nt < 4; ++nt)
      sP[(wv * 16 + r + 8 * half) * NTOK + nt * 16 + ln] = (_Float16)(lt[nt][r] * inv);
  }
  // sP rows read below belong to this wave only: per-wave LDS ordering suffices.

  v8f oacc[16];
  for (int nt = 0; nt < 16; ++nt) oacc[nt] = vzero();
#pragma unroll
  for (int ks = 0; ks < 64; ks += 32) {
    v16h a = loadA(sP + (wv * 16 + ln) * NTOK, ks, half);
#pragma unroll
    for (int nt = 0; nt < 16; ++nt)
      oacc[nt] = wmma16x16x32(a, loadB(sVt + (nt * 16 + ln) * NTOK, ks, half), oacc[nt]);
  }

  _Float16* qgB  = qg  + (size_t)(b * HEADS + head) * Lg * Dd;
  _Float16* qgTB = qgT + (size_t)(b * HEADS + head) * Dd * Lg;
  const int row0 = win * NTOK + wv * 16 + 8 * half;
#pragma unroll
  for (int nt = 0; nt < 16; ++nt) {
    h8 o8;
#pragma unroll
    for (int r = 0; r < 8; ++r) o8[r] = (_Float16)oacc[nt][r];
    st8(qgTB + (size_t)(nt * 16 + ln) * Lg + row0, o8);           // transposed: b128
#pragma unroll
    for (int r = 0; r < 8; ++r)                                    // row-major: scalar
      qgB[(size_t)(row0 + r) * Dd + nt * 16 + ln] = o8[r];
  }
}

// ---------------------------------------------------------------------------
// K3: global flash attention over L=4096 (Q=K=V=qg). 64 query rows per WG.
// All operands K-contiguous (qg row-major for S, qgT for PV). LDS: 8 KB.
// ---------------------------------------------------------------------------
__global__ void global_attn_kernel(const _Float16* __restrict__ qg,
                                   const _Float16* __restrict__ qgT,
                                   float* __restrict__ ogT) {
  __shared__ _Float16 sP[64 * 64];      // 8 KB
  const int tid = threadIdx.x;
  const int wv = tid >> 5, lane = tid & 31, half = lane >> 4, ln = lane & 15;
  const int blk = blockIdx.x, head = blockIdx.y, b = blockIdx.z;
  const float scale = 0.0625f;
  const _Float16* qgB  = qg  + (size_t)(b * HEADS + head) * Lg * Dd;
  const _Float16* qgTB = qgT + (size_t)(b * HEADS + head) * Dd * Lg;
  const _Float16* qrowp = qgB + (size_t)(blk * 64 + wv * 16 + ln) * Dd;

  v16h areg[8];                          // this lane's Q row, resident in VGPRs
#pragma unroll
  for (int kt = 0; kt < 8; ++kt) areg[kt] = loadA(qrowp, kt * 32, half);

  float mi[8], li[8];
  for (int r = 0; r < 8; ++r) { mi[r] = -3.0e38f; li[r] = 0.f; }
  v8f oacc[16];
  for (int nt = 0; nt < 16; ++nt) oacc[nt] = vzero();

  for (int cb = 0; cb < 64; ++cb) {
    // warm next K/V block into cache (global_prefetch_b8)
    __builtin_prefetch(qgB + (size_t)(((cb + 1) & 63) * 64) * Dd + tid * 128, 0, 1);

    v8f sx[4];
    for (int nt = 0; nt < 4; ++nt) sx[nt] = vzero();
    for (int kt = 0; kt < 8; ++kt) {
      const int kk = kt * 32;
#pragma unroll
      for (int nt = 0; nt < 4; ++nt)
        sx[nt] = wmma16x16x32(
            areg[kt], loadB(qgB + (size_t)(cb * 64 + nt * 16 + ln) * Dd, kk, half), sx[nt]);
    }

    // online softmax update (rows r+8*half per lane; reduce over 16-lane half)
    float p[4][8], alpha[8];
#pragma unroll
    for (int r = 0; r < 8; ++r) {
      float bmx = -3.0e38f;
      for (int nt = 0; nt < 4; ++nt) { p[nt][r] = sx[nt][r] * scale; bmx = fmaxf(bmx, p[nt][r]); }
      for (int off = 1; off < 16; off <<= 1) bmx = fmaxf(bmx, __shfl_xor(bmx, off, 32));
      float mn = fmaxf(mi[r], bmx);
      alpha[r] = __expf(mi[r] - mn);
      float ps = 0.f;
      for (int nt = 0; nt < 4; ++nt) { p[nt][r] = __expf(p[nt][r] - mn); ps += p[nt][r]; }
      for (int off = 1; off < 16; off <<= 1) ps += __shfl_xor(ps, off, 32);
      li[r] = li[r] * alpha[r] + ps;
      mi[r] = mn;
    }
#pragma unroll
    for (int nt = 0; nt < 16; ++nt)
#pragma unroll
      for (int r = 0; r < 8; ++r) oacc[nt][r] *= alpha[r];
#pragma unroll
    for (int r = 0; r < 8; ++r)
      for (int nt = 0; nt < 4; ++nt)
        sP[(wv * 16 + r + 8 * half) * 64 + nt * 16 + ln] = (_Float16)p[nt][r];
    // sP region is wave-private; per-wave LDS ordering suffices (no barrier)

#pragma unroll
    for (int ks = 0; ks < 64; ks += 32) {
      v16h a = loadA(sP + (wv * 16 + ln) * 64, ks, half);
#pragma unroll
      for (int nt = 0; nt < 16; ++nt)
        oacc[nt] = wmma16x16x32(
            a, loadB(qgTB + (size_t)(nt * 16 + ln) * Lg + cb * 64, ks, half), oacc[nt]);
    }
  }

  float* oT = ogT + (size_t)(b * HEADS + head) * Dd * Lg;
  const int row0 = blk * 64 + wv * 16 + 8 * half;
#pragma unroll
  for (int nt = 0; nt < 16; ++nt) {
    f4 lo, hi;
#pragma unroll
    for (int r = 0; r < 4; ++r) { lo[r] = oacc[nt][r] / li[r]; hi[r] = oacc[nt][r + 4] / li[r + 4]; }
    *(f4*)(oT + (size_t)(nt * 16 + ln) * Lg + row0)     = lo;
    *(f4*)(oT + (size_t)(nt * 16 + ln) * Lg + row0 + 4) = hi;
  }
}

// ---------------------------------------------------------------------------
// K4: overlap-add with analytic counts; reads transposed ogT[d][l] (4B stride).
// ---------------------------------------------------------------------------
__global__ void gather_avg_kernel(const float* __restrict__ ogT,
                                  float* __restrict__ ocrop) {
  int idx = blockIdx.x * blockDim.x + threadIdx.x;
  int b   = idx / (Cc * NPIX);
  int rem = idx % (Cc * NPIX);
  int ch  = rem / NPIX, pos = rem % NPIX;
  int r = (pos >> 3) + 3, c = (pos & 7) + 3;      // padded coords (crop offset 3)
  int head = ch >> 8, dim = ch & 255;
  int wilo = r - 7 < 0 ? 0 : r - 7, wihi = r < 7 ? r : 7;
  int wjlo = c - 7 < 0 ? 0 : c - 7, wjhi = c < 7 ? c : 7;
  const float* ogB = ogT + ((size_t)(b * HEADS + head) * Dd + dim) * Lg;
  float s = 0.f;
  for (int wi = wilo; wi <= wihi; ++wi)
    for (int wj = wjlo; wj <= wjhi; ++wj) {
      int tok = (r - wi) * 8 + (c - wj);
      s += ogB[(wi * 8 + wj) * NTOK + tok];
    }
  float cnt = (float)((wihi - wilo + 1) * (wjhi - wjlo + 1));
  ocrop[idx] = s / (cnt + 1e-6f);
}

// ---------------------------------------------------------------------------
// K5: final 1x1 projection + gated residual (a = 2*sigmoid(pa)).
// ---------------------------------------------------------------------------
__global__ void final_proj_kernel(const float* __restrict__ x,
                                  const float* __restrict__ Wp,
                                  const float* __restrict__ bp,
                                  const float* __restrict__ ocrop,
                                  const float* __restrict__ pa,
                                  float* __restrict__ out) {
  int idx = blockIdx.x * blockDim.x + threadIdx.x;
  int b   = idx / (Cc * NPIX);
  int rem = idx % (Cc * NPIX);
  int o   = rem / NPIX, p = rem % NPIX;
  float a = 2.0f / (1.0f + __expf(-pa[0]));
  float acc = bp[o];
  const float* ob = ocrop + (size_t)b * Cc * NPIX + p;
  const float* wr = Wp + (size_t)o * Cc;
  for (int c = 0; c < Cc; ++c) acc += wr[c] * ob[c * NPIX];
  out[idx] = x[idx] + a * acc;
}

// ---------------------------------------------------------------------------
// Host-side launcher.
// ---------------------------------------------------------------------------
extern "C" void kernel_launch(void* const* d_in, const int* in_sizes, int n_in,
                              void* d_out, int out_size, void* d_ws, size_t ws_size,
                              hipStream_t stream) {
  const float* x_feat = (const float*)d_in[0];
  const float* z_feat = (const float*)d_in[1];
  const float* Wqkv   = (const float*)d_in[2];
  const float* bqkv   = (const float*)d_in[3];
  const float* Wqkz   = (const float*)d_in[4];
  const float* bqkz   = (const float*)d_in[5];
  const float* Wproj  = (const float*)d_in[6];
  const float* bproj  = (const float*)d_in[7];
  const float* pa     = (const float*)d_in[8];
  const float* pbeta  = (const float*)d_in[9];
  float* out = (float*)d_out;
  (void)in_sizes; (void)n_in; (void)out_size; (void)ws_size;

  // workspace layout (~33.1 MB)
  char* w = (char*)d_ws;
  size_t off = 0;
  _Float16* xt   = (_Float16*)(w + off); off += (size_t)Bn * NPIX * Cc * 2;          // 128 KB
  _Float16* zt   = (_Float16*)(w + off); off += (size_t)Bn * NPIX * Cc * 2;          // 128 KB
  _Float16* qkvx = (_Float16*)(w + off); off += (size_t)Bn * NPIX * 3 * Cc * 2;      // 384 KB
  _Float16* qkz  = (_Float16*)(w + off); off += (size_t)Bn * NPIX * 2 * Cc * 2;      // 256 KB
  _Float16* qg   = (_Float16*)(w + off); off += (size_t)Bn * HEADS * Lg * Dd * 2;    // 8 MB
  _Float16* qgT  = (_Float16*)(w + off); off += (size_t)Bn * HEADS * Dd * Lg * 2;    // 8 MB
  float*    ogT  = (float*)(w + off);    off += (size_t)Bn * HEADS * Dd * Lg * 4;    // 16 MB
  float*    ocrp = (float*)(w + off);    off += (size_t)Bn * Cc * NPIX * 4;          // 256 KB

  const int NELT = Bn * Cc * NPIX;  // 65536
  to_f16_tr_kernel<<<NELT / 256, 256, 0, stream>>>(x_feat, xt);
  to_f16_tr_kernel<<<NELT / 256, 256, 0, stream>>>(z_feat, zt);
  proj_gemm_kernel<<<dim3(3 * Cc / 16, Bn), 128, 0, stream>>>(Wqkv, bqkv, xt, qkvx, 3 * Cc);
  proj_gemm_kernel<<<dim3(2 * Cc / 16, Bn), 128, 0, stream>>>(Wqkz, bqkz, zt, qkz, 2 * Cc);
  local_attn_kernel<<<dim3(NWIN, HEADS, Bn), 128, 0, stream>>>(qkvx, qkz, qg, qgT, pbeta);
  global_attn_kernel<<<dim3(Lg / 64, HEADS, Bn), 128, 0, stream>>>(qg, qgT, ogT);
  gather_avg_kernel<<<NELT / 256, 256, 0, stream>>>(ogT, ocrp);
  final_proj_kernel<<<NELT / 256, 256, 0, stream>>>(x_feat, Wproj, bproj, ocrp, pa, out);
}
